// BiLSTMForTwoTokenClassificationWithCRF_57586921504784
// MI455X (gfx1250) — matrix operand order repaired and checked by reading, not compiled, not run on hardware
//
#include <hip/hip_runtime.h>
#include <hip/hip_bf16.h>

#define BB   512
#define SS   128
#define DDIM 300
#define DPAD 320
#define HHID 256
#define GATE 1024   // 4*H
#define NLAB 9
#define NSE  5

typedef __bf16 bf16;
typedef bf16 v16bf __attribute__((ext_vector_type(16)));
typedef bf16 v8bf  __attribute__((ext_vector_type(8)));
typedef float v8f  __attribute__((ext_vector_type(8)));
typedef int   i32x4 __attribute__((vector_size(16)));

#define AS1 __attribute__((address_space(1)))
#define AS3 __attribute__((address_space(3)))

#if defined(__has_builtin)
#  if __has_builtin(__builtin_amdgcn_global_load_async_to_lds_b128)
#    define HAVE_ASYNC_LDS 1
#  else
#    define HAVE_ASYNC_LDS 0
#  endif
#  if __has_builtin(__builtin_amdgcn_s_wait_asynccnt)
#    define HAVE_WAIT_ASYNC 1
#  else
#    define HAVE_WAIT_ASYNC 0
#  endif
#else
#  define HAVE_ASYNC_LDS 0
#  define HAVE_WAIT_ASYNC 0
#endif

// ---------------------------------------------------------------------------
// WMMA fragment loaders (CDNA5 16x16x32 bf16 layout, ISA 7.12.2)
// A: lane(0..15)=row M, h=lane>>4 selects K-half; per lane two 16B chunks at
//    K = k0+8h and K = k0+16+8h  (VGPR0-3 / VGPR4-7)
// B: lane(0..15)=col N, same K order as A: chunks at K = k0+8h and K = k0+16+8h
// ---------------------------------------------------------------------------
__device__ __forceinline__ v16bf load_a_frag(const bf16* __restrict__ base, int ld,
                                             int row, int k0, int lane) {
    int h = lane >> 4;
    const bf16* p = base + (size_t)row * ld + k0 + 8 * h;
    v8bf c0 = *(const v8bf*)p;
    v8bf c1 = *(const v8bf*)(p + 16);
    v16bf r;
#pragma unroll
    for (int i = 0; i < 8; ++i) { r[i] = c0[i]; r[i + 8] = c1[i]; }
    return r;
}

__device__ __forceinline__ v16bf load_b_frag_lds(const bf16* lds, int ldsld,
                                                 int lane, int k0) {
    const bf16* p = lds + (lane & 15) * ldsld + k0 + 8 * (lane >> 4);
    v8bf c0 = *(const v8bf*)p;
    v8bf c1 = *(const v8bf*)(p + 16);
    v16bf r;
#pragma unroll
    for (int i = 0; i < 8; ++i) { r[i] = c0[i]; r[i + 8] = c1[i]; }
    return r;
}

// ---------------------------------------------------------------------------
// Cooperative staging of a 16-column weight tile (full K) into LDS.
// Uses GLOBAL_LOAD_ASYNC_TO_LDS_B128 (ASYNCcnt) when available.
// ---------------------------------------------------------------------------
__device__ __forceinline__ void stage_b_tile(const bf16* __restrict__ gsrc, int ld,
                                             int n0, int kelems, bf16* lds, int ldsld) {
    int chunksPerRow = kelems >> 3;            // 16B (8 bf16) chunks per row
    int total = chunksPerRow << 4;             // 16 rows
    for (int c = threadIdx.x; c < total; c += blockDim.x) {
        int row = c / chunksPerRow;
        int kk  = (c - row * chunksPerRow) << 3;
        const bf16* g = gsrc + (size_t)(n0 + row) * ld + kk;
        bf16* l = lds + row * ldsld + kk;
#if HAVE_ASYNC_LDS
        __builtin_amdgcn_global_load_async_to_lds_b128((AS1 i32x4*)g, (AS3 i32x4*)l, 0, 0);
#else
        *(v8bf*)l = *(const v8bf*)g;
#endif
    }
#if HAVE_ASYNC_LDS
#  if HAVE_WAIT_ASYNC
    __builtin_amdgcn_s_wait_asynccnt(0);
#  else
    asm volatile("s_wait_asynccnt 0x0" ::: "memory");
#  endif
#endif
    __syncthreads();
}

// ---------------------------------------------------------------------------
// Conversion / padding: f32 (rows x cols) -> bf16 (rows x colsp), zero padded
// ---------------------------------------------------------------------------
__global__ void conv_pad_bf16(const float* __restrict__ src, bf16* __restrict__ dst,
                              int rows, int cols, int colsp) {
    long idx = (long)blockIdx.x * blockDim.x + threadIdx.x;
    if (idx >= (long)rows * colsp) return;
    int r = (int)(idx / colsp), c = (int)(idx % colsp);
    dst[idx] = (c < cols) ? (bf16)src[(size_t)r * cols + c] : (bf16)0.0f;
}

// ---------------------------------------------------------------------------
// GEMM1: XG[M=B*S, N=GATE] = Xbf[M, DPAD] @ Wihᵀ + (b_ih + b_hh)
// block = 8 waves = 128M x 16N super-tile; B tile staged in LDS (async)
// ---------------------------------------------------------------------------
#define XG_LDSLD 328   // 320 + 8 pad -> conflict-free 16-lane column reads
__global__ void gemm_xg_kernel(const bf16* __restrict__ X, const bf16* __restrict__ W,
                               const float* __restrict__ bih, const float* __restrict__ bhh,
                               bf16* __restrict__ XG) {
    __shared__ bf16 sB[16 * XG_LDSLD];
    int bn = blockIdx.x & 63;          // N / 16 = 64 tiles
    int bm = blockIdx.x >> 6;          // 512 M-groups of 128 rows
    int n0 = bn * 16;
    stage_b_tile(W, DPAD, n0, DPAD, sB, XG_LDSLD);

    int wv = threadIdx.x >> 5, lane = threadIdx.x & 31;
    int m0 = bm * 128 + wv * 16;
    int h = lane >> 4;
    int rowA = m0 + (lane & 15);
    int colB = n0 + (lane & 15);
    v8f acc = {0.f, 0.f, 0.f, 0.f, 0.f, 0.f, 0.f, 0.f};
#pragma unroll 2
    for (int k0 = 0; k0 < DPAD; k0 += 32) {
        if (k0 + 32 < DPAD)
            __builtin_prefetch(X + (size_t)rowA * DPAD + k0 + 32, 0, 1);
        v16bf a = load_a_frag(X, DPAD, rowA, k0, lane);
        v16bf b = load_b_frag_lds(sB, XG_LDSLD, lane, k0);
        acc = __builtin_amdgcn_wmma_f32_16x16x32_bf16(false, a, false, b,
                                                      (short)0, acc, false, false);
    }
    float bias = bih[colB] + bhh[colB];
    bf16* outp = XG + (size_t)(m0 + 8 * h) * GATE + colB;
#pragma unroll
    for (int r = 0; r < 8; ++r) outp[(size_t)r * GATE] = (bf16)(acc[r] + bias);
}

// ---------------------------------------------------------------------------
// Per-step recurrent GEMM: Gbuf[B, GATE] = Hst[B, H] @ Whhᵀ + XG[:, t, :]
// block = 128M x 16N; Whh tile staged in LDS (async)
// ---------------------------------------------------------------------------
#define ST_LDSLD 264   // 256 + 8 pad
__global__ void gemm_step_kernel(const bf16* __restrict__ Hst, const bf16* __restrict__ Whh,
                                 const bf16* __restrict__ XG, int t,
                                 float* __restrict__ Gbuf) {
    __shared__ bf16 sB[16 * ST_LDSLD];
    int bn = blockIdx.x & 63;          // 64 N tiles
    int bm = blockIdx.x >> 6;          // 4 M-groups of 128 batch rows
    int n0 = bn * 16;
    stage_b_tile(Whh, HHID, n0, HHID, sB, ST_LDSLD);

    int wv = threadIdx.x >> 5, lane = threadIdx.x & 31;
    int m0 = bm * 128 + wv * 16;
    int h = lane >> 4;
    int rowA = m0 + (lane & 15);
    int colB = n0 + (lane & 15);
    v8f acc = {0.f, 0.f, 0.f, 0.f, 0.f, 0.f, 0.f, 0.f};
#pragma unroll
    for (int k0 = 0; k0 < HHID; k0 += 32) {
        v16bf a = load_a_frag(Hst, HHID, rowA, k0, lane);
        v16bf b = load_b_frag_lds(sB, ST_LDSLD, lane, k0);
        acc = __builtin_amdgcn_wmma_f32_16x16x32_bf16(false, a, false, b,
                                                      (short)0, acc, false, false);
    }
#pragma unroll
    for (int r = 0; r < 8; ++r) {
        int m = m0 + r + 8 * h;                                    // batch index
        float xv = (float)XG[((size_t)m * SS + t) * GATE + colB];  // precomputed xg
        Gbuf[(size_t)m * GATE + colB] = acc[r] + xv;
    }
}

// ---------------------------------------------------------------------------
// Pointwise LSTM cell update
// ---------------------------------------------------------------------------
__global__ void lstm_point_kernel(const float* __restrict__ Gbuf, float* __restrict__ Cst,
                                  bf16* __restrict__ Hst, bf16* __restrict__ HS, int t) {
    int idx = blockIdx.x * blockDim.x + threadIdx.x;   // B*H exact
    int b = idx >> 8, u = idx & (HHID - 1);
    const float* g = Gbuf + (size_t)b * GATE;
    float gi = g[u], gf = g[HHID + u], gg = g[2 * HHID + u], go = g[3 * HHID + u];
    float c  = Cst[idx];
    float si = 1.f / (1.f + __expf(-gi));
    float sf = 1.f / (1.f + __expf(-gf));
    float so = 1.f / (1.f + __expf(-go));
    c = sf * c + si * tanhf(gg);
    float hh = so * tanhf(c);
    Cst[idx] = c;
    Hst[idx] = (bf16)hh;
    HS[((size_t)b * SS + t) * HHID + u] = (bf16)hh;
}

__global__ void zero_state_kernel(float* __restrict__ Cst, bf16* __restrict__ Hst) {
    int idx = blockIdx.x * blockDim.x + threadIdx.x;
    Cst[idx] = 0.f;
    Hst[idx] = (bf16)0.f;
}

__global__ void zero_accum_kernel(float* __restrict__ acc) {
    if (threadIdx.x < 4) acc[threadIdx.x] = 0.f;
}

// ---------------------------------------------------------------------------
// Classifier: logits = [h_f | h_b] @ w_clsᵀ + b ;  se_logits likewise
// ---------------------------------------------------------------------------
__global__ void classifier_kernel(const bf16* __restrict__ HF, const bf16* __restrict__ HB,
                                  const float* __restrict__ Wc, const float* __restrict__ bc,
                                  const float* __restrict__ Wse, const float* __restrict__ bse,
                                  float* __restrict__ logits, float* __restrict__ selogits) {
    int idx = blockIdx.x * blockDim.x + threadIdx.x;
    int pos = idx / (NLAB + NSE);
    int j   = idx % (NLAB + NSE);
    if (pos >= BB * SS) return;
    const bf16* hf = HF + (size_t)pos * HHID;
    const bf16* hb = HB + (size_t)pos * HHID;
    const float* w; float s;
    if (j < NLAB) { w = Wc  + (size_t)j * (2 * HHID);          s = bc[j]; }
    else          { w = Wse + (size_t)(j - NLAB) * (2 * HHID); s = bse[j - NLAB]; }
#pragma unroll 4
    for (int u = 0; u < HHID; ++u)
        s += (float)hf[u] * w[u] + (float)hb[u] * w[HHID + u];
    if (j < NLAB) logits[(size_t)pos * NLAB + j] = s;
    else          selogits[(size_t)pos * NSE + (j - NLAB)] = s;
}

// ---------------------------------------------------------------------------
// CRF forward (loss numerator + partition) : one thread per batch row
// ---------------------------------------------------------------------------
template <int NT>
__global__ void crf_forward_kernel(const float* __restrict__ em, const int* __restrict__ tags,
                                   const int* __restrict__ mask, const float* __restrict__ startv,
                                   const float* __restrict__ endv, const float* __restrict__ trans,
                                   float* __restrict__ sum_llh, float* __restrict__ sum_mask,
                                   int do_mask) {
    __shared__ float str[NT * NT], sst[NT], sen[NT];
    for (int i = threadIdx.x; i < NT * NT; i += blockDim.x) str[i] = trans[i];
    for (int i = threadIdx.x; i < NT; i += blockDim.x) { sst[i] = startv[i]; sen[i] = endv[i]; }
    __syncthreads();
    int b = blockIdx.x * blockDim.x + threadIdx.x;
    if (b >= BB) return;
    const float* E  = em   + (size_t)b * SS * NT;
    const int*   tg = tags + (size_t)b * SS;
    const int*   mk = mask + (size_t)b * SS;
    float alpha[NT];
    int t0 = tg[0];
#pragma unroll
    for (int j = 0; j < NT; ++j) alpha[j] = sst[j] + E[j];
    float num = sst[t0] + E[t0];
    int prev = t0;
    int cnt = mk[0];
    for (int t = 1; t < SS; ++t) {
        int m = mk[t];
        int tt = tg[t];
        const float* Et = E + (size_t)t * NT;
        if (m) num += str[prev * NT + tt] + Et[tt];
        prev = tt;
        cnt += (m != 0);
        float na[NT];
#pragma unroll
        for (int j = 0; j < NT; ++j) {
            float mx = alpha[0] + str[j];
#pragma unroll
            for (int i = 1; i < NT; ++i) mx = fmaxf(mx, alpha[i] + str[i * NT + j]);
            float se = 0.f;
#pragma unroll
            for (int i = 0; i < NT; ++i) se += __expf(alpha[i] + str[i * NT + j] - mx);
            na[j] = mx + __logf(se) + Et[j];
        }
        if (m) {
#pragma unroll
            for (int j = 0; j < NT; ++j) alpha[j] = na[j];
        }
    }
    num += sen[tg[cnt - 1]];
    float mx = alpha[0] + sen[0];
#pragma unroll
    for (int j = 1; j < NT; ++j) mx = fmaxf(mx, alpha[j] + sen[j]);
    float se = 0.f;
#pragma unroll
    for (int j = 0; j < NT; ++j) se += __expf(alpha[j] + sen[j] - mx);
    float denom = mx + __logf(se);
    atomicAdd(sum_llh, num - denom);
    if (do_mask) atomicAdd(sum_mask, (float)cnt);
}

// ---------------------------------------------------------------------------
// Viterbi decode : one thread per batch row; backpointers in u8 workspace
// ---------------------------------------------------------------------------
template <int NT>
__global__ void viterbi_kernel(const float* __restrict__ em, const int* __restrict__ mask,
                               const float* __restrict__ startv, const float* __restrict__ endv,
                               const float* __restrict__ trans, unsigned char* __restrict__ hist,
                               float* __restrict__ outTags) {
    __shared__ float str[NT * NT], sst[NT], sen[NT];
    for (int i = threadIdx.x; i < NT * NT; i += blockDim.x) str[i] = trans[i];
    for (int i = threadIdx.x; i < NT; i += blockDim.x) { sst[i] = startv[i]; sen[i] = endv[i]; }
    __syncthreads();
    int b = blockIdx.x * blockDim.x + threadIdx.x;
    if (b >= BB) return;
    const float* E  = em + (size_t)b * SS * NT;
    const int*   mk = mask + (size_t)b * SS;
    unsigned char* hb_ = hist + (size_t)b * (SS - 1) * NT;
    float sc[NT];
#pragma unroll
    for (int j = 0; j < NT; ++j) sc[j] = sst[j] + E[j];
    for (int t = 1; t < SS; ++t) {
        int m = mk[t];
        const float* Et = E + (size_t)t * NT;
        float ns[NT];
#pragma unroll
        for (int j = 0; j < NT; ++j) {
            float best = sc[0] + str[j];
            int bi = 0;
#pragma unroll
            for (int i = 1; i < NT; ++i) {
                float v = sc[i] + str[i * NT + j];
                if (v > best) { best = v; bi = i; }
            }
            hb_[(size_t)(t - 1) * NT + j] = (unsigned char)bi;
            ns[j] = best + Et[j];
        }
        if (m) {
#pragma unroll
            for (int j = 0; j < NT; ++j) sc[j] = ns[j];
        }
    }
    float best = sc[0] + sen[0];
    int last = 0;
#pragma unroll
    for (int j = 1; j < NT; ++j) {
        float v = sc[j] + sen[j];
        if (v > best) { best = v; last = j; }
    }
    outTags[(size_t)b * SS + SS - 1] = (float)last;
    int cur = last;
    for (int t = SS - 2; t >= 0; --t) {
        if (mk[t + 1]) cur = hb_[(size_t)t * NT + cur];
        outTags[(size_t)b * SS + t] = (float)cur;
    }
}

__global__ void finalize_kernel(const float* __restrict__ acc, float* __restrict__ out) {
    if (threadIdx.x == 0) out[0] = -(acc[0] + acc[1]) / acc[2];
}

// ---------------------------------------------------------------------------
// Host-side orchestration
// ---------------------------------------------------------------------------
extern "C" void kernel_launch(void* const* d_in, const int* in_sizes, int n_in,
                              void* d_out, int out_size, void* d_ws, size_t ws_size,
                              hipStream_t stream) {
    const float* x       = (const float*)d_in[0];
    const int*   amask   = (const int*)  d_in[1];
    const int*   labels  = (const int*)  d_in[2];
    const int*   selab   = (const int*)  d_in[3];
    const float* w_ih_f  = (const float*)d_in[4];
    const float* w_hh_f  = (const float*)d_in[5];
    const float* b_ih_f  = (const float*)d_in[6];
    const float* b_hh_f  = (const float*)d_in[7];
    const float* w_ih_b  = (const float*)d_in[8];
    const float* w_hh_b  = (const float*)d_in[9];
    const float* b_ih_b  = (const float*)d_in[10];
    const float* b_hh_b  = (const float*)d_in[11];
    const float* w_cls   = (const float*)d_in[12];
    const float* b_cls   = (const float*)d_in[13];
    const float* w_se    = (const float*)d_in[14];
    const float* b_se    = (const float*)d_in[15];
    const float* start_t = (const float*)d_in[16];
    const float* end_t   = (const float*)d_in[17];
    const float* trans_t = (const float*)d_in[18];
    const float* start_s = (const float*)d_in[19];
    const float* end_s   = (const float*)d_in[20];
    const float* trans_s = (const float*)d_in[21];

    char* wsb = (char*)d_ws;
    size_t off = 0;
    auto take = [&](size_t bytes) { size_t o = off; off += (bytes + 255) & ~(size_t)255; return o; };
    bf16*  XBF  = (bf16*)(wsb + take((size_t)BB * SS * DPAD * 2));
    bf16*  WIHF = (bf16*)(wsb + take((size_t)GATE * DPAD * 2));
    bf16*  WIHB = (bf16*)(wsb + take((size_t)GATE * DPAD * 2));
    bf16*  WHHF = (bf16*)(wsb + take((size_t)GATE * HHID * 2));
    bf16*  WHHB = (bf16*)(wsb + take((size_t)GATE * HHID * 2));
    bf16*  XG   = (bf16*)(wsb + take((size_t)BB * SS * GATE * 2));   // reused per dir
    float* GBUF = (float*)(wsb + take((size_t)BB * GATE * 4));
    float* CST  = (float*)(wsb + take((size_t)BB * HHID * 4));
    bf16*  HST  = (bf16*)(wsb + take((size_t)BB * HHID * 2));
    bf16*  HF   = (bf16*)(wsb + take((size_t)BB * SS * HHID * 2));
    bf16*  HB   = (bf16*)(wsb + take((size_t)BB * SS * HHID * 2));
    unsigned char* HISTT = (unsigned char*)(wsb + take((size_t)BB * (SS - 1) * NLAB));
    unsigned char* HISTS = (unsigned char*)(wsb + take((size_t)BB * (SS - 1) * NSE));
    float* ACC  = (float*)(wsb + take(256));

    float* out        = (float*)d_out;
    float* out_logits = out + 1;
    float* out_se     = out + 1 + (size_t)BB * SS * NLAB;
    float* out_tags   = out + 1 + (size_t)BB * SS * (NLAB + NSE);
    float* out_setags = out_tags + (size_t)BB * SS;

    // 1. Convert / pad inputs and weights to bf16
    conv_pad_bf16<<<(BB * SS * DPAD) / 256, 256, 0, stream>>>(x, XBF, BB * SS, DDIM, DPAD);
    conv_pad_bf16<<<(GATE * DPAD) / 256, 256, 0, stream>>>(w_ih_f, WIHF, GATE, DDIM, DPAD);
    conv_pad_bf16<<<(GATE * DPAD) / 256, 256, 0, stream>>>(w_ih_b, WIHB, GATE, DDIM, DPAD);
    conv_pad_bf16<<<(GATE * HHID) / 256, 256, 0, stream>>>(w_hh_f, WHHF, GATE, HHID, HHID);
    conv_pad_bf16<<<(GATE * HHID) / 256, 256, 0, stream>>>(w_hh_b, WHHB, GATE, HHID, HHID);

    const int gemm1_blocks = (BB * SS / 128) * (GATE / 16);   // 32768
    const int step_blocks  = (BB / 128) * (GATE / 16);        // 256
    const int pt_blocks    = (BB * HHID) / 256;               // 512

    // 2. Forward direction
    gemm_xg_kernel<<<gemm1_blocks, 256, 0, stream>>>(XBF, WIHF, b_ih_f, b_hh_f, XG);
    zero_state_kernel<<<pt_blocks, 256, 0, stream>>>(CST, HST);
    for (int t = 0; t < SS; ++t) {
        gemm_step_kernel<<<step_blocks, 256, 0, stream>>>(HST, WHHF, XG, t, GBUF);
        lstm_point_kernel<<<pt_blocks, 256, 0, stream>>>(GBUF, CST, HST, HF, t);
    }

    // 3. Backward direction (reuses XG buffer)
    gemm_xg_kernel<<<gemm1_blocks, 256, 0, stream>>>(XBF, WIHB, b_ih_b, b_hh_b, XG);
    zero_state_kernel<<<pt_blocks, 256, 0, stream>>>(CST, HST);
    for (int t = SS - 1; t >= 0; --t) {
        gemm_step_kernel<<<step_blocks, 256, 0, stream>>>(HST, WHHB, XG, t, GBUF);
        lstm_point_kernel<<<pt_blocks, 256, 0, stream>>>(GBUF, CST, HST, HB, t);
    }

    // 4. Classifier heads -> d_out logits regions
    classifier_kernel<<<(BB * SS * (NLAB + NSE)) / 256, 256, 0, stream>>>(
        HF, HB, w_cls, b_cls, w_se, b_se, out_logits, out_se);

    // 5. CRF losses
    zero_accum_kernel<<<1, 32, 0, stream>>>(ACC);
    crf_forward_kernel<NLAB><<<(BB + 255) / 256, 256, 0, stream>>>(
        out_logits, labels, amask, start_t, end_t, trans_t, &ACC[0], &ACC[2], 1);
    crf_forward_kernel<NSE><<<(BB + 255) / 256, 256, 0, stream>>>(
        out_se, selab, amask, start_s, end_s, trans_s, &ACC[1], &ACC[2], 0);
    finalize_kernel<<<1, 32, 0, stream>>>(ACC, out);

    // 6. Viterbi decodes
    viterbi_kernel<NLAB><<<(BB + 255) / 256, 256, 0, stream>>>(
        out_logits, amask, start_t, end_t, trans_t, HISTT, out_tags);
    viterbi_kernel<NSE><<<(BB + 255) / 256, 256, 0, stream>>>(
        out_se, amask, start_s, end_s, trans_s, HISTS, out_setags);
}
